// SuperpixelSAN_10582799417882
// MI455X (gfx1250) — matrix operand order repaired and checked by reading, not compiled, not run on hardware
//
#include <hip/hip_runtime.h>
#include <hip/hip_bf16.h>

// ---------------------------------------------------------------------------
// Superpixel SAN forward on gfx1250 (MI455X).
//  - Dense linears (x@W+b): v_wmma_f32_16x16x32_f16 (f16 in, f32 acc),
//    templated on (DIN,DOUT,LDX,LDY) so all bounds and addressing are
//    compile-time; loads are unconditional (clamped addr + cndmask select).
//  - Sparse GAT softmax-aggregation: 3 edge passes with encoded-float
//    atomicMax, exp/segment-sum, and f32 atomicAdd scatter (bandwidth-bound).
// ---------------------------------------------------------------------------

typedef __attribute__((ext_vector_type(16))) _Float16 v16h;
typedef __attribute__((ext_vector_type(8)))  float    v8f;

#define NB 1024          // batch graphs
#define MAXN 786432
#define MAXNNZ 4718592

static __device__ __forceinline__ float leaky(float x) {
    return x > 0.f ? x : 0.01f * x;
}

// ---------------- fill ----------------
__global__ void fill_f32(float* __restrict__ p, float v, int n) {
    int i = blockIdx.x * blockDim.x + threadIdx.x;
    if (i < n) p[i] = v;
}
__global__ void fill_u32(unsigned* __restrict__ p, unsigned v, int n) {
    int i = blockIdx.x * blockDim.x + threadIdx.x;
    if (i < n) p[i] = v;
}

// ---------------- WMMA linear: Y[n x DOUT] = X[n x DIN] @ W + bias ----------
// One wave per 16-row tile; all strides compile-time; loads unconditional.
template<int DIN, int DOUT, int LDX, int LDY>
__global__ void linear_wmma(const float* __restrict__ X,
                            const float* __restrict__ W,
                            const float* __restrict__ bias,
                            float* __restrict__ Y, int n) {
    constexpr int NT = (DOUT + 15) / 16;   // 1 or 2
    constexpr int KT = (DIN + 31) / 32;    // 1 or 2
    int wave = threadIdx.x >> 5;
    int lane = threadIdx.x & 31;
    int tile = blockIdx.x * (blockDim.x >> 5) + wave;
    int row0 = tile * 16;
    if (row0 >= n) return;                 // wave-uniform: EXEC stays full
    int m = lane & 15, g = lane >> 4;

    int  r   = row0 + m;
    bool rok = (r < n);
    int  rc  = rok ? r : (n - 1);          // clamped, always valid address
    const float* xr = X + (size_t)rc * LDX;
    int ncolA = lane & 15;

    v8f c[NT];
#pragma unroll
    for (int j = 0; j < NT; ++j) c[j] = (v8f){};

#pragma unroll
    for (int t = 0; t < KT; ++t) {
        const int kbase = t * 32;
        // A tile: 16x32 f16; element i of lane: k = kbase+(i&7)+8g+16*(i>>3)
        v16h a;
#pragma unroll
        for (int i = 0; i < 16; ++i) {
            int  k  = kbase + (i & 7) + 8 * g + 16 * (i >> 3);
            bool ok = rok && (k < DIN);
            int  kc = (k < DIN) ? k : 0;
            float xv = xr[kc];             // unconditional load
            a[i] = (_Float16)(ok ? xv : 0.f);
        }
        // B tiles: element i of lane: k = kbase+(i&7)+8g+16*(i>>3), col = j*16+(lane&15)
#pragma unroll
        for (int j = 0; j < NT; ++j) {
            v16h bm;
            int ncol = j * 16 + ncolA;
            int nc   = (ncol < DOUT) ? ncol : (DOUT - 1);
#pragma unroll
            for (int i = 0; i < 16; ++i) {
                int  k  = kbase + (i & 7) + 8 * g + 16 * (i >> 3);
                int  kc = (k < DIN) ? k : 0;
                float wv = W[kc * DOUT + nc];           // unconditional load
                bool ok  = (k < DIN) && (ncol < DOUT);
                bm[i] = (_Float16)(ok ? wv : 0.f);
            }
            c[j] = __builtin_amdgcn_wmma_f32_16x16x32_f16(false, a, false, bm,
                                                          (short)0, c[j], false, false);
        }
    }
    // bias: one clamped unconditional load per N-tile (hoisted out of stores)
    float bs[NT];
#pragma unroll
    for (int j = 0; j < NT; ++j) {
        int ncol = j * 16 + ncolA;
        bs[j] = bias[(ncol < DOUT) ? ncol : (DOUT - 1)];
    }
    // store: C/D VGPR rr holds row row0+rr+8g, col = j*16 + (lane&15)
    float* yb = Y + (size_t)(row0 + 8 * g) * LDY;
#pragma unroll
    for (int rr = 0; rr < 8; ++rr) {
        int rw = row0 + rr + 8 * g;
        if (rw >= n) continue;
#pragma unroll
        for (int j = 0; j < NT; ++j) {
            int ncol = j * 16 + ncolA;
            if (ncol < DOUT)
                yb[rr * LDY + ncol] = c[j][rr] + bs[j];
        }
    }
}

// ---------------- per-node attention scalars ----------------
__global__ void attn_score(const float* __restrict__ h,
                           const float* __restrict__ a1, const float* __restrict__ a1b,
                           const float* __restrict__ a2, const float* __restrict__ a2b,
                           float* __restrict__ s1, float* __restrict__ s2, int n) {
    int i = blockIdx.x * blockDim.x + threadIdx.x;
    if (i >= n) return;
    const float* hi = h + (size_t)i * 18;
    float d1 = 0.f, d2 = 0.f;
#pragma unroll
    for (int f = 0; f < 18; ++f) {
        float v = hi[f];
        d1 += v * a1[f];
        d2 += v * a2[f];
    }
    s1[i] = d1 + a1b[0];
    s2[i] = d2 + a2b[0];
}

// ---------------- edge pass 1: logits + segment max (encoded atomicMax) -----
__global__ void edge_logit_max(const int* __restrict__ row, const int* __restrict__ col,
                               const float* __restrict__ s1, const float* __restrict__ s2,
                               unsigned* __restrict__ menc, int nnz) {
    int i = blockIdx.x * blockDim.x + threadIdx.x;
    if (i >= nnz) return;
    int r = row[i];
    float e = leaky(s1[r] + s2[col[i]]);
    unsigned u = __float_as_uint(e);
    u = (u & 0x80000000u) ? ~u : (u | 0x80000000u);   // order-preserving encode
    atomicMax(&menc[r], u);
}

// ---------------- edge pass 2: exp(e - m), segment sum ----------------------
__global__ void edge_exp_sum(const int* __restrict__ row, const int* __restrict__ col,
                             const float* __restrict__ s1, const float* __restrict__ s2,
                             const unsigned* __restrict__ menc,
                             float* __restrict__ ex, float* __restrict__ den, int nnz) {
    int i = blockIdx.x * blockDim.x + threadIdx.x;
    if (i >= nnz) return;
    int r = row[i];
    float e = leaky(s1[r] + s2[col[i]]);
    unsigned u = menc[r];
    float mval = (u & 0x80000000u) ? __uint_as_float(u ^ 0x80000000u)
                                   : __uint_as_float(~u);
    float v = __expf(e - mval);
    ex[i] = v;
    atomicAdd(&den[r], v);
}

// ---------------- edge pass 3: acc[row] += (ex/den[row]) * h[col] -----------
__global__ void edge_aggregate(const int* __restrict__ row, const int* __restrict__ col,
                               const float* __restrict__ ex, const float* __restrict__ den,
                               const float* __restrict__ h, float* __restrict__ acc, int nnz) {
    int i = blockIdx.x * blockDim.x + threadIdx.x;
    if (i >= nnz) return;
    int r = row[i], c = col[i];
    float att = ex[i] / den[r];
    const float* hc = h + (size_t)c * 18;
    float* ar = acc + (size_t)r * 18;
#pragma unroll
    for (int f = 0; f < 18; ++f) atomicAdd(&ar[f], att * hc[f]);
}

// ---------------- SpMM: acc[row] += val * h[col] ----------------------------
__global__ void spmm_edge(const int* __restrict__ row, const int* __restrict__ col,
                          const float* __restrict__ val,
                          const float* __restrict__ h, float* __restrict__ acc, int nnz) {
    int i = blockIdx.x * blockDim.x + threadIdx.x;
    if (i >= nnz) return;
    int r = row[i], c = col[i];
    float v = val[i];
    const float* hc = h + (size_t)c * 18;
    float* ar = acc + (size_t)r * 18;
#pragma unroll
    for (int f = 0; f < 18; ++f) atomicAdd(&ar[f], v * hc[f]);
}

// ---------------- relu + copy into 54-wide concat buffer --------------------
__global__ void relu_copy(const float* __restrict__ acc, float* __restrict__ H,
                          int n, int colOff) {
    int idx = blockIdx.x * blockDim.x + threadIdx.x;
    if (idx >= n * 18) return;
    int i = idx / 18, f = idx % 18;
    float v = acc[idx];
    H[(size_t)i * 54 + colOff + f] = v > 0.f ? v : 0.f;
}

// ---------------- global mean pool (accumulate) -----------------------------
__global__ void pool_accum(const float* __restrict__ h4, const int* __restrict__ batch,
                           float* __restrict__ P, float* __restrict__ cnt,
                           int n, int branch) {
    int i = blockIdx.x * blockDim.x + threadIdx.x;
    if (i >= n) return;
    int b = batch[i];
    const float* hi = h4 + (size_t)i * 10;
    float* pb = P + (size_t)b * 30 + branch * 10;
#pragma unroll
    for (int f = 0; f < 10; ++f) atomicAdd(&pb[f], hi[f]);
    atomicAdd(&cnt[branch * NB + b], 1.0f);
}

__global__ void pool_norm(float* __restrict__ P, const float* __restrict__ cnt) {
    int idx = blockIdx.x * blockDim.x + threadIdx.x;
    if (idx >= NB * 30) return;
    int b = idx / 30, col = idx % 30;
    int br = col / 10;
    P[idx] /= fmaxf(cnt[br * NB + b], 1.0f);
}

// ---------------- row softmax over 10 classes -------------------------------
__global__ void softmax10(const float* __restrict__ logits, float* __restrict__ out) {
    int b = blockIdx.x * blockDim.x + threadIdx.x;
    if (b >= NB) return;
    const float* l = logits + (size_t)b * 10;
    float mx = l[0];
#pragma unroll
    for (int f = 1; f < 10; ++f) mx = fmaxf(mx, l[f]);
    float e[10], s = 0.f;
#pragma unroll
    for (int f = 0; f < 10; ++f) { e[f] = __expf(l[f] - mx); s += e[f]; }
    float inv = 1.f / s;
#pragma unroll
    for (int f = 0; f < 10; ++f) out[(size_t)b * 10 + f] = e[f] * inv;
}

// ===========================================================================
static inline int cdiv(int a, int b) { return (a + b - 1) / b; }

// shape ids: 0:<3,18,3,18> 1:<6,18,6,18> 2:<9,18,9,18> 3:<18,18,54,18>
//            4:<54,10,54,10> 5:<30,10,30,10>
static void launch_linear(int shape, const float* X,
                          const float* W, const float* b,
                          float* Y, int n, hipStream_t stream) {
    dim3 grid(cdiv(n, 128)), block(256);   // 8 waves x 16 rows per block
    switch (shape) {
        case 0: linear_wmma<3, 18, 3, 18><<<grid, block, 0, stream>>>(X, W, b, Y, n); break;
        case 1: linear_wmma<6, 18, 6, 18><<<grid, block, 0, stream>>>(X, W, b, Y, n); break;
        case 2: linear_wmma<9, 18, 9, 18><<<grid, block, 0, stream>>>(X, W, b, Y, n); break;
        case 3: linear_wmma<18, 18, 54, 18><<<grid, block, 0, stream>>>(X, W, b, Y, n); break;
        case 4: linear_wmma<54, 10, 54, 10><<<grid, block, 0, stream>>>(X, W, b, Y, n); break;
        default: linear_wmma<30, 10, 30, 10><<<grid, block, 0, stream>>>(X, W, b, Y, n); break;
    }
}

extern "C" void kernel_launch(void* const* d_in, const int* in_sizes, int n_in,
                              void* d_out, int out_size, void* d_ws, size_t ws_size,
                              hipStream_t stream) {
    (void)in_sizes; (void)n_in; (void)out_size; (void)ws_size;
    auto F = [&](int i) { return (const float*)d_in[i]; };
    auto I = [&](int i) { return (const int*)d_in[i]; };

    // ---- workspace bump layout (floats) ----
    float* ws = (float*)d_ws;
    float*    H    = ws;                       // MAXN*54 concat(h1,h2,h3)
    float*    acc  = H   + (size_t)MAXN * 54;  // MAXN*18 (reused as h4: n*10)
    float*    h    = acc + (size_t)MAXN * 18;  // MAXN*18 (hp / hu / hd, reused)
    float*    s1   = h   + (size_t)MAXN * 18;  // MAXN
    float*    s2   = s1  + MAXN;               // MAXN
    unsigned* menc = (unsigned*)(s2 + MAXN);   // MAXN
    float*    den  = (float*)menc + MAXN;      // MAXN
    float*    ex   = den + MAXN;               // MAXNNZ
    float*    P    = ex  + MAXNNZ;             // NB*30 pooled concat
    float*    cnt  = P   + NB * 30;            // 3*NB
    float*    lg   = cnt + 3 * NB;             // NB*10 logits

    const int TB = 256;

    // param leaf indices (top-level dict in insertion order; nested params
    // pytree-flattened: dict keys sorted, lists in order).
    // 19: comb.W  20: comb.b  21/22: lin0.W/b  23/24: lin1  25/26: lin2
    // 27+: o0 (3 layers x 14 leaves), 69+: o1, 111+: o2
    // per layer: dW,da1,da1b,da2,da2b,db, pW,pb, uW,ua1,ua1b,ua2,ua2b,ub
    struct Br {
        const float* X; int n, shape0;
        const int *prow, *pcol; const float* pval; int pnnz;
        const int *urow, *ucol; int unnz;
        const int *drow, *dcol; int dnnz;
        const int* batch; int oBase, linW;
    };
    Br br[3] = {
        { F(0), 262144, 0, I(3),  I(4),  F(5),  1835008,
          nullptr, nullptr, 0,  I(3),  I(4),  1835008, I(16), 27,  21 },
        { F(1), 786432, 1, I(6),  I(7),  F(8),  4718592,
          I(9),  I(10), 2359296, I(11), I(12), 2359296, I(17), 69,  23 },
        { F(2), 393216, 2, I(13), I(14), F(15), 2359296,
          I(13), I(14), 2359296, nullptr, nullptr, 0,   I(18), 111, 25 },
    };

    // zero pooled sums + counts
    fill_f32<<<cdiv(NB * 30, TB), TB, 0, stream>>>(P, 0.f, NB * 30);
    fill_f32<<<cdiv(3 * NB, TB), TB, 0, stream>>>(cnt, 0.f, 3 * NB);

    for (int bidx = 0; bidx < 3; ++bidx) {
        const Br& b = br[bidx];
        const int n = b.n;
        const int gN   = cdiv(n, TB);
        const int gN18 = cdiv(n * 18, TB);

        for (int l = 0; l < 3; ++l) {
            int pl = b.oBase + 14 * l;
            const float* inX; int shape;
            if (l == 0) { inX = b.X;              shape = b.shape0; }
            else        { inX = H + 18 * (l - 1); shape = 3; }   // col offset folded

            fill_f32<<<gN18, TB, 0, stream>>>(acc, 0.f, n * 18);

            // ---- p path: acc += spmm(p, x@pW + pb) ----
            launch_linear(shape, inX, F(pl + 6), F(pl + 7), h, n, stream);
            spmm_edge<<<cdiv(b.pnnz, TB), TB, 0, stream>>>(b.prow, b.pcol, b.pval,
                                                           h, acc, b.pnnz);
            // ---- u / d GAT paths ----
            for (int gd = 0; gd < 2; ++gd) {
                const int* erow = gd ? b.drow : b.urow;
                const int* ecol = gd ? b.dcol : b.ucol;
                int        nnz  = gd ? b.dnnz : b.unnz;
                int        pg   = gd ? pl + 0 : pl + 8;   // d leaves at +0, u at +8
                if (nnz == 0) continue;
                launch_linear(shape, inX, F(pg + 0), F(pg + 5), h, n, stream);
                attn_score<<<gN, TB, 0, stream>>>(h, F(pg + 1), F(pg + 2),
                                                  F(pg + 3), F(pg + 4), s1, s2, n);
                fill_u32<<<gN, TB, 0, stream>>>(menc, 0u, n);
                fill_f32<<<gN, TB, 0, stream>>>(den, 0.f, n);
                int gE = cdiv(nnz, TB);
                edge_logit_max<<<gE, TB, 0, stream>>>(erow, ecol, s1, s2, menc, nnz);
                edge_exp_sum<<<gE, TB, 0, stream>>>(erow, ecol, s1, s2, menc, ex, den, nnz);
                edge_aggregate<<<gE, TB, 0, stream>>>(erow, ecol, ex, den, h, acc, nnz);
            }
            relu_copy<<<gN18, TB, 0, stream>>>(acc, H, n, 18 * l);
        }
        // h4 = concat(h1,h2,h3) @ linW + linb  (reuse acc as [n x 10])
        launch_linear(4, H, F(b.linW), F(b.linW + 1), acc, n, stream);
        pool_accum<<<gN, TB, 0, stream>>>(acc, b.batch, P, cnt, n, bidx);
    }

    pool_norm<<<cdiv(NB * 30, TB), TB, 0, stream>>>(P, cnt);
    launch_linear(5, P, F(19), F(20), lg, NB, stream);
    softmax10<<<cdiv(NB, TB), TB, 0, stream>>>(lg, (float*)d_out);
}
